// TemporalGCN_5849745457230
// MI455X (gfx1250) — matrix operand order repaired and checked by reading, not compiled, not run on hardware
//
#include <hip/hip_runtime.h>
#include <hip/hip_bf16.h>
#include <cstdint>
#include <cstddef>

// ---------------------------------------------------------------------------
// TemporalGCN for MI455X (gfx1250), wave32 + WMMA bf16 + global_prefetch.
//   B=16, N=400, T=96, F=1, H=128, O=8
// Pipeline:
//   K1 rowsum -> d[400]
//   K2 build na (bf16, K padded to 416) + naT (f32)
//   K3 weight prep: gc2_w / w_ih / w_hh -> pre-swizzled WMMA B-frag layouts
//   K4 gc1 (F=1 algebraic shortcut): p[bt][m] = (na @ xm) * combined
//   K5 GEMM1 (fused h1): A-frags built in-register from p, z stored directly
//      in B-fragment global layout (zf) with b128 stores        (WMMA)
//   K6 GEMM2: per (b,t) na @ z; B-frags = direct coalesced global loads from
//      zf; prefetch next slab; epilogue *mask + gc2_b           (WMMA)
//   K7 persistent GRU: gx & gh per step via WMMA (B-frags in registers),
//      gates in LDS, prefetch x_{t+1}, split FC head + mask     (WMMA)
// ---------------------------------------------------------------------------

typedef __attribute__((ext_vector_type(16))) __bf16 v16bf;
typedef __attribute__((ext_vector_type(8)))  __bf16 v8bf;
typedef __attribute__((ext_vector_type(8)))  float  v8f;

#define WMMA_BF16(a, b, c) \
  __builtin_amdgcn_wmma_f32_16x16x32_bf16(false, (a), false, (b), (short)0, (c), false, false)

constexpr int Bb = 16, Nn = 400, Tt = 96, Hh = 128, Oo = 8;
constexpr int NP = 416;              // node dim padded to 13*32 for WMMA K
constexpr int BT = Bb * Tt;          // 1536
constexpr int RR = BT * NP;          // padded rows of h1/z = 638976
constexpr int MSEQ = Bb * Nn;        // 6400 GRU sequences
constexpr int ZFBT = NP * Hh;        // 53248 elems per (b,t) slab of zf

__device__ __forceinline__ v8f vzero8() {
  v8f v = {0.f, 0.f, 0.f, 0.f, 0.f, 0.f, 0.f, 0.f};
  return v;
}

// A-matrix fragment (16x32 bf16), contiguous K layout: lane m=lane&15, hi=lane>>4;
// elems 0..15 -> K = kOff+hi*16+e
// => two contiguous 16B chunks per lane (ds_load_b128 x2).
__device__ __forceinline__ v16bf load_afrag(const __bf16* base, int rowStride, int kOff) {
  int lane = threadIdx.x & 31;
  int m = lane & 15, hi = lane >> 4;
  const __bf16* p = base + m * rowStride + kOff + hi * 16;
  v8bf lo = *(const v8bf*)p;
  v8bf hv = *(const v8bf*)(p + 8);
  return __builtin_shufflevector(lo, hv, 0, 1, 2, 3, 4, 5, 6, 7, 8, 9, 10, 11, 12, 13, 14, 15);
}

// ---------------- K1: degree rsqrt -------------------------------------------
__global__ void k_rowsum(const float* __restrict__ adj, float* __restrict__ d) {
  __shared__ float red[128];
  int m = blockIdx.x;
  float s = 0.f;
  for (int n = threadIdx.x; n < Nn; n += 128) s += adj[m * Nn + n];
  red[threadIdx.x] = s;
  __syncthreads();
  for (int w = 64; w > 0; w >>= 1) {
    if (threadIdx.x < w) red[threadIdx.x] += red[threadIdx.x + w];
    __syncthreads();
  }
  if (threadIdx.x == 0) {
    float t = red[0] + 1.0f;  // +I on diagonal
    d[m] = (t > 0.f) ? rsqrtf(t) : 0.f;
  }
}

// ---------------- K2: na (bf16 padded) + naT (f32) ---------------------------
__global__ void k_build_na(const float* __restrict__ adj, const float* __restrict__ d,
                           __bf16* __restrict__ nab, float* __restrict__ naT) {
  int idx = blockIdx.x * 256 + threadIdx.x;
  if (idx >= Nn * NP) return;
  int m = idx / NP, k = idx % NP;
  float v = 0.f;
  if (k < Nn) {
    v = (adj[m * Nn + k] + (m == k ? 1.f : 0.f)) * d[m] * d[k];
    naT[k * Nn + m] = v;
  }
  nab[m * NP + k] = (__bf16)v;
}

// ---------------- K3a: gc2_w [128][128] -> B-frag layout ---------------------
// gc2wf[((s*8+ct)*32+L)*16+e] = w[k=s*32+e+16*(L>=16)][n=ct*16+(L&15)]
__global__ void k_swz_gc2w(const float* __restrict__ w, __bf16* __restrict__ wf) {
  int idx = blockIdx.x * 256 + threadIdx.x;
  if (idx >= 4 * 8 * 32 * 16) return;
  int e = idx & 15, L = (idx >> 4) & 31, ct = (idx >> 9) & 7, s = idx >> 12;
  int k = s * 32 + e + ((L >= 16) ? 16 : 0);
  int n = ct * 16 + (L & 15);
  wf[idx] = (__bf16)w[k * Hh + n];
}

// ---------------- K3b: GRU weight [384][128] -> B-frag layout ----------------
// wf[((c*4+s)*32+L)*16+e] = w[g=c*16+(L&15)][k=s*32+e+16*(L>=16)]
__global__ void k_swz(const float* __restrict__ w, __bf16* __restrict__ wf) {
  int idx = blockIdx.x * 256 + threadIdx.x;
  if (idx >= 24 * 4 * 32 * 16) return;
  int e = idx & 15, L = (idx >> 4) & 31, s = (idx >> 9) & 3, c = idx >> 11;
  int g = c * 16 + (L & 15);
  int k = s * 32 + e + ((L >= 16) ? 16 : 0);
  wf[idx] = (__bf16)w[g * Hh + k];
}

// ---------------- K4: gc1 adjacency, fused combined --------------------------
// p[bt][m] = (sum_n na[m][n]*xm[b][n][t]) * missing[b][m][t] * mask[b][m][t]
__global__ void k_gc1(const float* __restrict__ x, const float* __restrict__ mask,
                      const float* __restrict__ naT, float* __restrict__ p) {
  __shared__ float xs[Nn];
  __shared__ float cs[Nn];
  int bt = blockIdx.x;
  int b = bt / Tt, t = bt % Tt;
  int tid = threadIdx.x;
  if (tid < Nn) {
    float v = x[(b * Nn + tid) * Tt + t];
    float miss = (v != -1.0f) ? 1.f : 0.f;
    xs[tid] = v * miss;
    cs[tid] = miss * mask[(b * Nn + tid) * Tt + t];
  }
  __syncthreads();
  if (tid < Nn) {
    float acc = 0.f;
    for (int k = 0; k < Nn; ++k) acc += naT[k * Nn + tid] * xs[k];  // coalesced
    p[bt * Nn + tid] = acc * cs[tid];
  }
}

// ---------------- K5: GEMM1  z = relu(p*gc1w+gc1b) @ gc2_w -------------------
// h1 built on the fly in registers (per-row scalar p). Output goes straight
// to B-fragment layout zf, one global_store_b128 per col tile.
__global__ void k_gemm1(const float* __restrict__ p, const float* __restrict__ gc1w,
                        const float* __restrict__ gc1b, const __bf16* __restrict__ gc2wf,
                        __bf16* __restrict__ zf) {
  __shared__ float w1s[Hh];
  __shared__ float b1s[Hh];
  int tid = threadIdx.x;
  if (tid < Hh) { w1s[tid] = gc1w[tid]; b1s[tid] = gc1b[tid]; }
  __syncthreads();

  int wave = tid >> 5, lane = tid & 31;
  int nn = lane & 15, hi = lane >> 4;
  int rt = blockIdx.x * 8 + wave;          // 16-row tile of padded rows
  int bt = rt / 26, btile = rt % 26;       // 26 tiles per (b,t)
  int nrow = btile * 16 + nn;              // node index of this lane's row
  bool valid = (nrow < Nn);
  float pv = valid ? p[bt * Nn + nrow] : 0.f;

  v8f acc[8];
#pragma unroll
  for (int j = 0; j < 8; ++j) acc[j] = vzero8();

#pragma unroll
  for (int s = 0; s < 4; ++s) {
    // build A fragment in registers: relu(pv*w1[k]+b1[k]), zero for pad rows
    v16bf a;
#pragma unroll
    for (int e = 0; e < 16; ++e) {
      int k = s * 32 + hi * 16 + e;
      float vv = fmaxf(pv * w1s[k] + b1s[k], 0.f);
      a[e] = (__bf16)(valid ? vv : 0.f);
    }
#pragma unroll
    for (int ct = 0; ct < 8; ++ct) {
      v16bf b = *(const v16bf*)(gc2wf + (((s * 8 + ct) * 32 + lane) << 4));
      acc[ct] = WMMA_BF16(a, b, acc[ct]);
    }
  }

  // store C tile directly into zf (B-frag layout for GEMM2):
  // this tile's rows are K-values kl0..kl0+15 of slab sT, half hiP.
  int kl0 = btile * 16;
  int sT = kl0 >> 5;
  int hiP = (kl0 >> 4) & 1;
  __bf16* zb = zf + (size_t)bt * ZFBT;
#pragma unroll
  for (int ct = 0; ct < 8; ++ct) {
    v8bf pk;
#pragma unroll
    for (int r = 0; r < 8; ++r) pk[r] = (__bf16)acc[ct][r];
    *(v8bf*)(zb + (((sT * 8 + ct) * 32 + hiP * 16 + nn) << 4) + hi * 8) = pk;
  }
}

// ---------------- K6: GEMM2 adjacency per (b,t): na @ z ----------------------
// grid: BT*5 blocks; block = 5 m-tiles (80 rows), wave w -> h-tile w.
// B-frags: direct coalesced global v16bf loads from zf (no LDS for B).
// Next slab's A rows + B-frag prefetched (global_prefetch_b8).
__global__ void k_gemm2(const __bf16* __restrict__ nab, const __bf16* __restrict__ zf,
                        const float* __restrict__ mask, const float* __restrict__ gc2b,
                        __bf16* __restrict__ xg) {
  __shared__ __align__(16) __bf16 As[80 * 40];  // 80 na rows x 32 K, padded stride
  int blk = blockIdx.x;
  int bt = blk / 5, mg = blk % 5;
  int b = bt / Tt, t = bt % Tt;
  int m0 = mg * 80;
  int tid = threadIdx.x, wave = tid >> 5, lane = tid & 31;
  int nn = lane & 15, hi = lane >> 4;

  const __bf16* zb = zf + (size_t)bt * ZFBT;
  v8f acc[5];
#pragma unroll
  for (int j = 0; j < 5; ++j) acc[j] = vzero8();

  for (int s = 0; s < 13; ++s) {
    __syncthreads();  // previous slab consumed
    for (int i = tid; i < 1280; i += 256) {      // 80x32 bf16 as dwords
      int r = (i * 2) / 32, k = (i * 2) % 32;
      *(uint32_t*)&As[r * 40 + k] = *(const uint32_t*)&nab[(m0 + r) * NP + s * 32 + k];
    }
    if (s + 1 < 13) {
      // prefetch next A slab rows and this wave's next B-fragment
      if (tid < 80) __builtin_prefetch(&nab[(m0 + tid) * NP + (s + 1) * 32], 0, 1);
      __builtin_prefetch(zb + ((((s + 1) * 8 + wave) * 32 + lane) << 4), 0, 1);
    }
    __syncthreads();
    v16bf bf = *(const v16bf*)(zb + (((s * 8 + wave) * 32 + lane) << 4));
#pragma unroll
    for (int m5 = 0; m5 < 5; ++m5) {
      v16bf a = load_afrag(As + m5 * 16 * 40, 40, 0);
      acc[m5] = WMMA_BF16(a, bf, acc[m5]);
    }
  }

  // epilogue: *combined2 + gc2_b. After ReLU, h1 >= 0 so (h1 != -1) == 1,
  // hence combined2 == mask exactly.
  int hcol = wave * 16 + nn;
  float bias = gc2b[hcol];
#pragma unroll
  for (int m5 = 0; m5 < 5; ++m5)
#pragma unroll
    for (int r = 0; r < 8; ++r) {
      int m = m0 + m5 * 16 + r + hi * 8;  // < 400 always
      float mv = mask[(b * Nn + m) * Tt + t];
      float v = acc[m5][r] * mv + bias;
      xg[((size_t)t * MSEQ + b * Nn + m) * Hh + hcol] = (__bf16)v;  // [t][seq][h]
    }
}

// ---------------- K7: persistent GRU + fused FC head -------------------------
// 400 blocks x 512 threads (16 waves). 48 tile-jobs/step:
//   jobs 0..23 -> gx cols (B=w_ihT), 24..47 -> gh cols (B=w_hhT)
// B-fragments preloaded once into registers from pre-swizzled arrays.
__global__ void __launch_bounds__(512, 1)
k_gru(const __bf16* __restrict__ xg, const __bf16* __restrict__ wfih,
      const __bf16* __restrict__ wfhh, const float* __restrict__ bih,
      const float* __restrict__ bhh, const float* __restrict__ fcw,
      const float* __restrict__ fcb, const float* __restrict__ mask,
      float* __restrict__ out) {
  __shared__ float gX[16 * 384];                    // 24 KB (reused for FC partials)
  __shared__ float gH[16 * 384];                    // 24 KB
  __shared__ __align__(16) __bf16 h_s[16 * 136];    // hidden state (bf16)
  __shared__ __align__(16) __bf16 x_s[16 * 136];    // staged gru-input tile
  __shared__ float bb[768];
  __shared__ float fcw_s[Hh * Oo];
  __shared__ float fcb_s[Oo];

  int tid = threadIdx.x, wave = tid >> 5, lane = tid & 31;
  int seq0 = blockIdx.x * 16;

  if (tid < 384) { bb[tid] = bih[tid]; bb[384 + tid] = bhh[tid]; }
  for (int i = tid; i < Hh * Oo; i += 512) fcw_s[i] = fcw[i];
  if (tid < Oo) fcb_s[tid] = fcb[tid];
  for (int i = tid; i < 16 * 136 / 2; i += 512) ((uint32_t*)h_s)[i] = 0u;  // h0 = 0

  v16bf bfr[3][4];
#pragma unroll
  for (int i = 0; i < 3; ++i) {
    int j = wave + i * 16;
    const __bf16* wf = (j >= 24) ? wfhh : wfih;
    int c = j % 24;
#pragma unroll
    for (int s = 0; s < 4; ++s)
      bfr[i][s] = *(const v16bf*)(wf + (((c * 4 + s) * 32 + lane) << 4));
  }
  __syncthreads();

  for (int t = 0; t < Tt; ++t) {
    // stage x_t tile: contiguous 16x128 block thanks to [t][seq][h] layout
    {
      const uint32_t* src = (const uint32_t*)(xg + ((size_t)t * MSEQ + seq0) * Hh);
      for (int i = tid; i < 1024; i += 512) {
        int m = i >> 6, dw = i & 63;
        ((uint32_t*)&x_s[m * 136])[dw] = src[m * 64 + dw];
      }
      // prefetch next step's tile (4 KB) while this step computes
      if (t + 1 < Tt && tid < 256)
        __builtin_prefetch((const char*)(xg + ((size_t)(t + 1) * MSEQ + seq0) * Hh) + tid * 16,
                           0, 1);
    }
    __syncthreads();

    // two GEMMs per step: gx = x_t @ w_ih^T ; gh = h @ w_hh^T
#pragma unroll
    for (int i = 0; i < 3; ++i) {
      int j = wave + i * 16;
      int matB = (j >= 24);
      int c = j % 24;
      const __bf16* Asrc = matB ? h_s : x_s;
      v8f acc = vzero8();
#pragma unroll
      for (int s = 0; s < 4; ++s) {
        v16bf a = load_afrag(Asrc, 136, s * 32);
        acc = WMMA_BF16(a, bfr[i][s], acc);
      }
      float* gdst = matB ? gH : gX;
      int n = lane & 15, hi = lane >> 4;
#pragma unroll
      for (int r = 0; r < 8; ++r) gdst[(r + hi * 8) * 384 + c * 16 + n] = acc[r];
    }
    __syncthreads();

    // gate math + hidden update; two h columns per thread, packed b32 store
    for (int i = tid; i < 1024; i += 512) {
      int m = i >> 6, h2 = (i & 63) * 2;
      uint32_t hpk = *(const uint32_t*)&h_s[m * 136 + h2];
      float hnew[2];
#pragma unroll
      for (int j = 0; j < 2; ++j) {
        int h = h2 + j;
        float xr = gX[m * 384 + h] + bb[h];
        float hr = gH[m * 384 + h] + bb[384 + h];
        float xz = gX[m * 384 + 128 + h] + bb[128 + h];
        float hz = gH[m * 384 + 128 + h] + bb[384 + 128 + h];
        float xn = gX[m * 384 + 256 + h] + bb[256 + h];
        float hn = gH[m * 384 + 256 + h] + bb[384 + 256 + h];
        float r = 1.f / (1.f + __expf(-(xr + hr)));
        float zg = 1.f / (1.f + __expf(-(xz + hz)));
        float ng = tanhf(xn + r * hn);
        uint16_t hbits = (uint16_t)((hpk >> (16 * j)) & 0xFFFF);
        float hp = (float)*(__bf16*)&hbits;
        hnew[j] = (1.f - zg) * ng + zg * hp;
      }
      __bf16 p0 = (__bf16)hnew[0], p1 = (__bf16)hnew[1];
      uint32_t packed = (uint32_t)*(uint16_t*)&p0 | ((uint32_t)*(uint16_t*)&p1 << 16);
      *(uint32_t*)&h_s[m * 136 + h2] = packed;
    }
    __syncthreads();

    // fused FC head, all 512 threads: 4 h-chunks -> partials in gX (dead here)
    {
      int g = tid >> 7;              // 0..3: h-chunk
      int mo = tid & 127;
      int m = mo >> 3, o = mo & 7;
      float acc = 0.f;
#pragma unroll 4
      for (int h = g * 32; h < g * 32 + 32; ++h)
        acc += (float)h_s[m * 136 + h] * fcw_s[h * Oo + o];
      gX[g * 128 + mo] = acc;
    }
    __syncthreads();
    if (tid < 128) {
      int m = tid >> 3, o = tid & 7;
      float acc = fcb_s[o] + gX[tid] + gX[128 + tid] + gX[256 + tid] + gX[384 + tid];
      int seq = seq0 + m;
      out[((size_t)seq * Tt + t) * Oo + o] = acc * mask[seq * Tt + t];
    }
    __syncthreads();
  }
}

// ---------------- workspace layout -------------------------------------------
constexpr size_t AL(size_t x) { return (x + 255) & ~(size_t)255; }
constexpr size_t OFF_D    = 0;
constexpr size_t OFF_NAT  = AL(OFF_D + (size_t)Nn * 4);
constexpr size_t OFF_NAB  = AL(OFF_NAT + (size_t)Nn * Nn * 4);
constexpr size_t OFF_P    = AL(OFF_NAB + (size_t)Nn * NP * 2);
constexpr size_t OFF_G2WF = AL(OFF_P + (size_t)BT * Nn * 4);
constexpr size_t OFF_WFI  = AL(OFF_G2WF + (size_t)4 * 8 * 32 * 16 * 2);
constexpr size_t OFF_WFH  = AL(OFF_WFI + (size_t)49152 * 2);
constexpr size_t OFF_ZF   = AL(OFF_WFH + (size_t)49152 * 2);           // 163.6 MB
constexpr size_t OFF_XG   = AL(OFF_ZF + (size_t)RR * Hh * 2);          // 157.3 MB

extern "C" void kernel_launch(void* const* d_in, const int* in_sizes, int n_in,
                              void* d_out, int out_size, void* d_ws, size_t ws_size,
                              hipStream_t stream) {
  const float* x    = (const float*)d_in[0];
  const float* adj  = (const float*)d_in[1];
  const float* mask = (const float*)d_in[2];
  const float* gc1w = (const float*)d_in[3];
  const float* gc1b = (const float*)d_in[4];
  const float* gc2w = (const float*)d_in[5];
  const float* gc2b = (const float*)d_in[6];
  const float* wih  = (const float*)d_in[7];
  const float* whh  = (const float*)d_in[8];
  const float* bih  = (const float*)d_in[9];
  const float* bhh  = (const float*)d_in[10];
  const float* fcw  = (const float*)d_in[11];
  const float* fcb  = (const float*)d_in[12];
  float* out = (float*)d_out;
  char* ws = (char*)d_ws;

  float*  d_deg = (float*)(ws + OFF_D);
  float*  naT   = (float*)(ws + OFF_NAT);
  __bf16* nab   = (__bf16*)(ws + OFF_NAB);
  float*  pbuf  = (float*)(ws + OFF_P);
  __bf16* g2wf  = (__bf16*)(ws + OFF_G2WF);
  __bf16* wfih  = (__bf16*)(ws + OFF_WFI);
  __bf16* wfhh  = (__bf16*)(ws + OFF_WFH);
  __bf16* zf    = (__bf16*)(ws + OFF_ZF);
  __bf16* xg    = (__bf16*)(ws + OFF_XG);

  k_rowsum<<<Nn, 128, 0, stream>>>(adj, d_deg);
  k_build_na<<<(Nn * NP + 255) / 256, 256, 0, stream>>>(adj, d_deg, nab, naT);
  k_swz_gc2w<<<(16384 + 255) / 256, 256, 0, stream>>>(gc2w, g2wf);
  k_swz<<<192, 256, 0, stream>>>(wih, wfih);
  k_swz<<<192, 256, 0, stream>>>(whh, wfhh);
  k_gc1<<<BT, 512, 0, stream>>>(x, mask, naT, pbuf);
  k_gemm1<<<RR / 16 / 8, 256, 0, stream>>>(pbuf, gc1w, gc1b, g2wf, zf);
  k_gemm2<<<BT * 5, 256, 0, stream>>>(nab, zf, mask, gc2b, xg);
  k_gru<<<MSEQ / 16, 512, 0, stream>>>(xg, wfih, wfhh, bih, bhh, fcw, fcb, mask, out);
}
